// MultiScaleAttentionHighFre_3702261809534
// MI455X (gfx1250) — compile-verified, matrix operand order checked by reading, not run on hardware
//
#include <hip/hip_runtime.h>
#include <hip/hip_bf16.h>

typedef __attribute__((ext_vector_type(16))) _Float16 v16h;
typedef __attribute__((ext_vector_type(8)))  float    v8f;
typedef __attribute__((ext_vector_type(8)))  _Float16 v8h;
typedef unsigned int       u32;
typedef unsigned long long u64;

#define B_DIM   16
#define N_TOK   512
#define C_DIM   4096
#define H_NUM   8
#define HD      512
#define QKV_LD  12288

// ---------------- workspace layout (bytes) ----------------
static constexpr size_t OFF_QKV  = 0;                                     // f16 [8192][12288]
static constexpr size_t OFF_FW   = (size_t)8192 * 12288 * 2;              // f32 [16][512][4096]
static constexpr size_t OFF_YMID = OFF_FW   + (size_t)16 * 512 * 4096 * 4;// f16 [16][<=960][4096]
static constexpr size_t OFF_S    = OFF_YMID + (size_t)16 * 960 * 4096 * 2;// f32 [128][512][512]
static constexpr size_t OFF_P    = OFF_S    + (size_t)128 * 512 * 512 * 4;// f16 [128][512][512]
static constexpr size_t OFF_XH   = OFF_P    + (size_t)128 * 512 * 512 * 2;// f16 x
static constexpr size_t OFF_WQH  = OFF_XH   + (size_t)16 * 512 * 4096 * 2;// f16 qkv_w
static constexpr size_t OFF_WPH  = OFF_WQH  + (size_t)3 * 4096 * 4096 * 2;// f16 proj_w
// y2 (f32, <=252MB) aliases [0, OFF_YMID): qkv & freq_w are dead before proj GEMM.

// ---------------- CDNA5 primitives ----------------
__device__ __forceinline__ v8f wmma16(v16h a, v16h b, v8f c) {
  return __builtin_amdgcn_wmma_f32_16x16x32_f16(false, a, false, b, (short)0, c, false, false);
}

// Async 16B copy global -> LDS (tracked by ASYNCcnt). LDS byte address is the
// low 32 bits of the generic pointer (ISA 10.2).
__device__ __forceinline__ void async_b128(u32 lds_off, u64 gaddr) {
  asm volatile("global_load_async_to_lds_b128 %0, %1, off"
               :: "v"(lds_off), "v"(gaddr) : "memory");
}
__device__ __forceinline__ void wait_async0() {
  asm volatile("s_wait_asynccnt 0x0" ::: "memory");
}

// A/B fragment for v_wmma_f32_16x16x32_f16 from an LDS row (row-major over K).
// lanes 0-15: K = {0..7, 16..23}; lanes 16-31: K = {8..15, 24..31}.
__device__ __forceinline__ v16h frag_from_lds(const _Float16* row, int koff) {
  v16h r;
#pragma unroll
  for (int j = 0; j < 8; ++j) r[j] = row[koff + j];
#pragma unroll
  for (int j = 0; j < 8; ++j) r[8 + j] = row[16 + koff + j];
  return r;
}

__device__ __forceinline__ v8f zero8() {
  v8f z = {0.f, 0.f, 0.f, 0.f, 0.f, 0.f, 0.f, 0.f};
  return z;
}

// ================= K0: f32 -> f16 convert (one-time) =================
__global__ void cvt_f16_kernel(const float* __restrict__ src, _Float16* __restrict__ dst, int n) {
  const int i = (blockIdx.x * 256 + threadIdx.x) * 8;
  if (i + 8 > n) return;
  const float4 a = *(const float4*)(src + i);
  const float4 b = *(const float4*)(src + i + 4);
  v8h o;
  o[0] = (_Float16)a.x; o[1] = (_Float16)a.y; o[2] = (_Float16)a.z; o[3] = (_Float16)a.w;
  o[4] = (_Float16)b.x; o[5] = (_Float16)b.y; o[6] = (_Float16)b.z; o[7] = (_Float16)b.w;
  *(v8h*)(dst + i) = o;
}

// ================= K1: 512-pt FFT along tokens -> freq_w =================
__global__ void fft_freqw_kernel(const float* __restrict__ x, float* __restrict__ fw) {
  __shared__ float re[512];
  __shared__ float im[512];
  const int t = threadIdx.x;                 // 256 threads
  const int b = blockIdx.x >> 12;
  const int c = blockIdx.x & 4095;
  const float* col = x + (size_t)b * N_TOK * C_DIM + c;
#pragma unroll
  for (int i = t; i < 512; i += 256) {
    int rv = (int)(__brev((unsigned)i) >> 23);     // 9-bit bit reversal
    re[rv] = col[(size_t)i * C_DIM];
    im[rv] = 0.f;
  }
  __syncthreads();
  for (int len = 2; len <= 512; len <<= 1) {
    const int half = len >> 1;
    const int blk  = t / half;
    const int pos  = t - blk * half;
    const int i0 = blk * len + pos;
    const int i1 = i0 + half;
    float sn, cs;
    __sincosf(-6.28318530717958647692f * (float)pos / (float)len, &sn, &cs);
    const float xr = re[i1], xi = im[i1];
    const float tr = xr * cs - xi * sn;
    const float ti = xr * sn + xi * cs;
    const float ur = re[i0], ui = im[i0];
    re[i0] = ur + tr; im[i0] = ui + ti;
    re[i1] = ur - tr; im[i1] = ui - ti;
    __syncthreads();
  }
#pragma unroll
  for (int i = t; i < 512; i += 256) {
    const float amp = sqrtf(re[i] * re[i] + im[i] * im[i]);
    const float w = 1.f / (1.f + __expf(-2.0f * (amp - 0.5f)));
    fw[((size_t)b * N_TOK + i) * C_DIM + c] = w;
  }
}

// ========== big GEMM: C = A(f16,MxK) * B(f16,NxK)^T, async double-buffered ==========
// Block tile 128M x 256N x 32K; 8 waves, 64x64 per wave (4x4 WMMA accums).
template <bool OUT_HALF>
__global__ void gemm_hh_kernel(const _Float16* __restrict__ A, const _Float16* __restrict__ Bw,
                               const float* __restrict__ bias, void* __restrict__ Cout,
                               int lda, int ldb, int ldc, int K, float oscale) {
  __shared__ alignas(16) _Float16 As[2][128][40];
  __shared__ alignas(16) _Float16 Bs[2][256][40];
  const int tid = threadIdx.x;
  const int wave = tid >> 5, lane = tid & 31;
  const int wm = (wave >> 2) * 64, wn = (wave & 3) * 64;
  const int bm = blockIdx.y * 128, bn = blockIdx.x * 256;

  v8f acc[4][4];
#pragma unroll
  for (int i = 0; i < 4; ++i)
#pragma unroll
    for (int j = 0; j < 4; ++j) acc[i][j] = zero8();

  auto issueA = [&](int buf, int k0) {
#pragma unroll
    for (int i = 0; i < 2; ++i) {                    // 512 16B-chunks / 256 thr
      const int ch = tid + i * 256;
      const int row = ch >> 2, q = ch & 3;
      const u32 l = (u32)(uintptr_t)&As[buf][row][q * 8];
      const _Float16* g = A + (size_t)(bm + row) * lda + k0 + q * 8;
      async_b128(l, (u64)(uintptr_t)g);
    }
  };
  auto issueB = [&](int buf, int k0) {
#pragma unroll
    for (int i = 0; i < 4; ++i) {                    // 1024 16B-chunks / 256 thr
      const int ch = tid + i * 256;
      const int row = ch >> 2, q = ch & 3;
      const u32 l = (u32)(uintptr_t)&Bs[buf][row][q * 8];
      const _Float16* g = Bw + (size_t)(bn + row) * ldb + k0 + q * 8;
      async_b128(l, (u64)(uintptr_t)g);
    }
  };

  issueA(0, 0);
  issueB(0, 0);
  wait_async0();
  __syncthreads();

  const int lrow = lane & 15;
  const int koff = (lane < 16) ? 0 : 8;
  int buf = 0;
  for (int k0 = 0; k0 < K; k0 += 32) {
    const int nxt = buf ^ 1;
    const bool more = (k0 + 32) < K;
    if (more) { issueA(nxt, k0 + 32); issueB(nxt, k0 + 32); }

    v16h af[4], bf[4];
#pragma unroll
    for (int i = 0; i < 4; ++i) af[i] = frag_from_lds(&As[buf][wm + i * 16 + lrow][0], koff);
#pragma unroll
    for (int j = 0; j < 4; ++j) bf[j] = frag_from_lds(&Bs[buf][wn + j * 16 + lrow][0], koff);
#pragma unroll
    for (int i = 0; i < 4; ++i)
#pragma unroll
      for (int j = 0; j < 4; ++j) acc[i][j] = wmma16(af[i], bf[j], acc[i][j]);

    if (more) wait_async0();
    __syncthreads();
    buf = nxt;
  }

  const int nlo = lane & 15;
  const int mhi = (lane < 16) ? 0 : 8;
#pragma unroll
  for (int i = 0; i < 4; ++i)
#pragma unroll
    for (int j = 0; j < 4; ++j) {
      const int gn = bn + wn + j * 16 + nlo;
      const float bv = bias ? bias[gn] : 0.f;
#pragma unroll
      for (int r = 0; r < 8; ++r) {
        const int gm = bm + wm + i * 16 + mhi + r;
        const float v = (acc[i][j][r] + bv) * oscale;
        if (OUT_HALF) ((_Float16*)Cout)[(size_t)gm * ldc + gn] = (_Float16)v;
        else          ((float*)Cout)[(size_t)gm * ldc + gn] = v;
      }
    }
}

// ================= K3: S = Q_s K_s^T * sm_scale (per b,h, downsample stride s) =================
__global__ void attn_score_kernel(const _Float16* __restrict__ qkv, float* __restrict__ S,
                                  int s, int hd_s, float sm_scale) {
  __shared__ _Float16 Qs[128][40];
  __shared__ _Float16 Ks[128][40];
  const int tid = threadIdx.x;
  const int wave = tid >> 5, lane = tid & 31;
  const int wm = (wave >> 2) * 64, wn = (wave & 3) * 32;
  const int qt = (blockIdx.x & 3) * 128;
  const int kt = (blockIdx.x >> 2) * 128;
  const int bh = blockIdx.y, b = bh >> 3, h = bh & 7;
  const int lr = tid >> 1, lc = (tid & 1) * 16;
  const size_t qrow = (size_t)(b * N_TOK + qt + lr) * QKV_LD + h * HD;           // q slice
  const size_t krow = (size_t)(b * N_TOK + kt + lr) * QKV_LD + C_DIM + h * HD;   // k slice

  v8f acc[4][2];
#pragma unroll
  for (int i = 0; i < 4; ++i)
#pragma unroll
    for (int j = 0; j < 2; ++j) acc[i][j] = zero8();

  const int lrow = lane & 15;
  const int koff = (lane < 16) ? 0 : 8;

  for (int k0 = 0; k0 < hd_s; k0 += 32) {
#pragma unroll
    for (int j = 0; j < 16; ++j) Qs[lr][lc + j] = qkv[qrow + (size_t)(k0 + lc + j) * s];
#pragma unroll
    for (int j = 0; j < 16; ++j) Ks[lr][lc + j] = qkv[krow + (size_t)(k0 + lc + j) * s];
    __syncthreads();
    v16h af[4], bf[2];
#pragma unroll
    for (int i = 0; i < 4; ++i) af[i] = frag_from_lds(&Qs[wm + i * 16 + lrow][0], koff);
#pragma unroll
    for (int j = 0; j < 2; ++j) bf[j] = frag_from_lds(&Ks[wn + j * 16 + lrow][0], koff);
#pragma unroll
    for (int i = 0; i < 4; ++i)
#pragma unroll
      for (int j = 0; j < 2; ++j) acc[i][j] = wmma16(af[i], bf[j], acc[i][j]);
    __syncthreads();
  }

  const int nlo = lane & 15;
  const int mhi = (lane < 16) ? 0 : 8;
#pragma unroll
  for (int i = 0; i < 4; ++i)
#pragma unroll
    for (int j = 0; j < 2; ++j)
#pragma unroll
      for (int r = 0; r < 8; ++r) {
        const int gq = qt + wm + i * 16 + mhi + r;
        const int gk = kt + wn + j * 16 + nlo;
        S[((size_t)bh * N_TOK + gq) * N_TOK + gk] = acc[i][j][r] * sm_scale;
      }
}

// ================= K4: P = softmax(S) * freq_w, store f16 =================
__global__ void softmax_fw_kernel(const float* __restrict__ S, const float* __restrict__ FW,
                                  _Float16* __restrict__ P) {
  __shared__ float red[256];
  const int t = threadIdx.x;
  const int row = blockIdx.x;                 // bh*512 + q
  const int bh = row >> 9, q = row & 511;
  const int b = bh >> 3, h = bh & 7;
  const float* sr = S + (size_t)row * N_TOK;
  const float v0 = sr[t], v1 = sr[t + 256];
  red[t] = fmaxf(v0, v1);
  __syncthreads();
  for (int o = 128; o > 0; o >>= 1) { if (t < o) red[t] = fmaxf(red[t], red[t + o]); __syncthreads(); }
  const float mx = red[0];
  __syncthreads();
  const float e0 = __expf(v0 - mx), e1 = __expf(v1 - mx);
  red[t] = e0 + e1;
  __syncthreads();
  for (int o = 128; o > 0; o >>= 1) { if (t < o) red[t] += red[t + o]; __syncthreads(); }
  const float inv = 1.f / red[0];
  const float* fw = FW + (size_t)(b * N_TOK + q) * C_DIM + h * HD;  // hd == N == 512
  _Float16* pr = P + (size_t)row * N_TOK;
  pr[t]       = (_Float16)(e0 * inv * fw[t]);
  pr[t + 256] = (_Float16)(e1 * inv * fw[t + 256]);
}

// ================= K5: O = P @ V_s, scatter into y_mid (f16) =================
__global__ void attn_out_kernel(const _Float16* __restrict__ P, const _Float16* __restrict__ qkv,
                                _Float16* __restrict__ ymid, int s, int hd_s, int rowoff, int Mrows) {
  __shared__ _Float16 Ps[128][40];
  __shared__ _Float16 Vs[128][40];   // Vs[d][k] = V^T
  const int tid = threadIdx.x;
  const int wave = tid >> 5, lane = tid & 31;
  const int wm = (wave >> 2) * 64, wn = (wave & 3) * 32;
  const int qt = (blockIdx.x & 3) * 128;
  const int dt = (blockIdx.x >> 2) * 128;
  const int bh = blockIdx.y, b = bh >> 3, h = bh & 7;
  const int lr = tid >> 1, lc = (tid & 1) * 16;

  v8f acc[4][2];
#pragma unroll
  for (int i = 0; i < 4; ++i)
#pragma unroll
    for (int j = 0; j < 2; ++j) acc[i][j] = zero8();

  const int lrow = lane & 15;
  const int koff = (lane < 16) ? 0 : 8;
  const size_t vcol = (size_t)(2 * C_DIM) + h * HD + (size_t)(dt + lr) * s;

  for (int k0 = 0; k0 < N_TOK; k0 += 32) {
    const _Float16* pp = P + ((size_t)bh * N_TOK + qt + lr) * N_TOK + k0 + lc;
#pragma unroll
    for (int j = 0; j < 16; ++j) Ps[lr][lc + j] = pp[j];
#pragma unroll
    for (int j = 0; j < 16; ++j)
      Vs[lr][lc + j] = qkv[(size_t)(b * N_TOK + k0 + lc + j) * QKV_LD + vcol];
    __syncthreads();
    v16h af[4], bf[2];
#pragma unroll
    for (int i = 0; i < 4; ++i) af[i] = frag_from_lds(&Ps[wm + i * 16 + lrow][0], koff);
#pragma unroll
    for (int j = 0; j < 2; ++j) bf[j] = frag_from_lds(&Vs[wn + j * 16 + lrow][0], koff);
#pragma unroll
    for (int i = 0; i < 4; ++i)
#pragma unroll
      for (int j = 0; j < 2; ++j) acc[i][j] = wmma16(af[i], bf[j], acc[i][j]);
    __syncthreads();
  }

  const int nlo = lane & 15;
  const int mhi = (lane < 16) ? 0 : 8;
  const size_t base = (size_t)b * Mrows * C_DIM + (size_t)rowoff * C_DIM;
#pragma unroll
  for (int i = 0; i < 4; ++i)
#pragma unroll
    for (int j = 0; j < 2; ++j)
#pragma unroll
      for (int r = 0; r < 8; ++r) {
        const int gq = qt + wm + i * 16 + mhi + r;
        const int gd = dt + wn + j * 16 + nlo;
        ymid[base + (size_t)(h * N_TOK + gq) * hd_s + gd] = (_Float16)acc[i][j][r];
      }
}

// ================= K7: LayerNorm rows of y2 -> out =================
__global__ void layernorm_kernel(const float* __restrict__ Y, const float* __restrict__ g,
                                 const float* __restrict__ bta, float* __restrict__ out) {
  __shared__ float rs[256];
  __shared__ float rq[256];
  const int t = threadIdx.x;
  const size_t row = blockIdx.x;
  const float* y = Y + row * C_DIM;
  float s1 = 0.f, s2 = 0.f;
#pragma unroll
  for (int i = 0; i < 16; ++i) { const float v = y[t + i * 256]; s1 += v; s2 += v * v; }
  rs[t] = s1; rq[t] = s2;
  __syncthreads();
  for (int o = 128; o > 0; o >>= 1) {
    if (t < o) { rs[t] += rs[t + o]; rq[t] += rq[t + o]; }
    __syncthreads();
  }
  const float mu = rs[0] * (1.f / 4096.f);
  const float var = rq[0] * (1.f / 4096.f) - mu * mu;
  const float rstd = rsqrtf(var + 1e-5f);
  float* op = out + row * C_DIM;
#pragma unroll
  for (int i = 0; i < 16; ++i) {
    const int c = t + i * 256;
    op[c] = (y[c] - mu) * rstd * g[c] + bta[c];
  }
}

// ========================= host launch =========================
extern "C" void kernel_launch(void* const* d_in, const int* in_sizes, int n_in,
                              void* d_out, int out_size, void* d_ws, size_t ws_size,
                              hipStream_t stream) {
  const float* x      = (const float*)d_in[0];
  const float* qkv_w  = (const float*)d_in[1];
  const float* qkv_b  = (const float*)d_in[2];
  const float* proj_w = (const float*)d_in[3];
  const float* proj_b = (const float*)d_in[4];
  const float* ln_g   = (const float*)d_in[5];
  const float* ln_b   = (const float*)d_in[6];
  float* out = (float*)d_out;

  // n_scales is determined by output row count: Mrows = sum_i 512/2^i
  const int Mrows = out_size / (B_DIM * C_DIM);
  int n_scales;
  switch (Mrows) {
    case 512: n_scales = 1; break;
    case 768: n_scales = 2; break;
    case 960: n_scales = 4; break;
    default:  n_scales = 3; break;   // Mrows == 896
  }

  char* ws = (char*)d_ws;
  _Float16* qkv  = (_Float16*)(ws + OFF_QKV);
  float*    fw   = (float*)(ws + OFF_FW);
  _Float16* ymid = (_Float16*)(ws + OFF_YMID);
  float*    S    = (float*)(ws + OFF_S);
  _Float16* P    = (_Float16*)(ws + OFF_P);
  _Float16* xh   = (_Float16*)(ws + OFF_XH);
  _Float16* wqh  = (_Float16*)(ws + OFF_WQH);
  _Float16* wph  = (_Float16*)(ws + OFF_WPH);
  float*    y2   = (float*)ws;  // aliases qkv/fw region (both dead before proj GEMM)

  // K0: one-time f32 -> f16 conversion of GEMM operands
  const int n_x  = B_DIM * N_TOK * C_DIM;
  const int n_wq = 3 * C_DIM * C_DIM;
  const int n_wp = C_DIM * C_DIM;
  cvt_f16_kernel<<<n_x  / 2048, 256, 0, stream>>>(x,      xh,  n_x);
  cvt_f16_kernel<<<n_wq / 2048, 256, 0, stream>>>(qkv_w,  wqh, n_wq);
  cvt_f16_kernel<<<n_wp / 2048, 256, 0, stream>>>(proj_w, wph, n_wp);

  // K1: freq_w
  fft_freqw_kernel<<<B_DIM * C_DIM, 256, 0, stream>>>(x, fw);

  // K2: qkv = x @ qkv_w^T + qkv_b  (M=8192, N=12288, K=4096), f16 output
  gemm_hh_kernel<true><<<dim3(QKV_LD / 256, (B_DIM * N_TOK) / 128), 256, 0, stream>>>(
      xh, wqh, qkv_b, (void*)qkv, C_DIM, C_DIM, QKV_LD, C_DIM, 1.0f);

  // K3..K5 per scale
  const float sm_scale = 0.04419417382415922f;  // 512^-0.5 (always full hd)
  int rowoff = 0;
  for (int i = 0; i < n_scales; ++i) {
    const int s = 1 << i;
    const int hd_s = HD >> i;
    attn_score_kernel<<<dim3(16, B_DIM * H_NUM), 256, 0, stream>>>(qkv, S, s, hd_s, sm_scale);
    softmax_fw_kernel<<<B_DIM * H_NUM * N_TOK, 256, 0, stream>>>(S, fw, P);
    attn_out_kernel<<<dim3(4 * (hd_s / 128), B_DIM * H_NUM), 256, 0, stream>>>(
        P, qkv, ymid, s, hd_s, rowoff, Mrows);
    rowoff += N_TOK >> i;
  }

  // K6: y2 = 2*(ymid @ proj_w^T + proj_b)   (M=16*Mrows, N=4096, K=4096), f32 output
  const int Mtot = B_DIM * Mrows;
  gemm_hh_kernel<false><<<dim3(C_DIM / 256, Mtot / 128), 256, 0, stream>>>(
      ymid, wph, proj_b, (void*)y2, C_DIM, C_DIM, C_DIM, C_DIM, 2.0f);

  // K7: LayerNorm
  layernorm_kernel<<<Mtot, 256, 0, stream>>>(y2, ln_g, ln_b, out);
}
